// SelfBlock_86981677679341
// MI455X (gfx1250) — matrix operand order repaired and checked by reading, not compiled, not run on hardware
//
#include <hip/hip_runtime.h>
#include <cstdint>
#include <cstddef>

// ---------------------------------------------------------------------------
// MI455X / gfx1250 transformer block (QKV+RoPE+attention+out-proj+FFN+LN+GELU)
// All GEMMs and attention matmuls use v_wmma_f32_16x16x32_bf16 (bf16 in, f32
// accumulate). Elementwise math (softmax, LN, GELU, RoPE, bias, residual) in
// fp32. Compute-bound workload (~240 GFLOP vs ~0.5GB traffic @ 23.3 TB/s).
// GEMM: 128x256 block tile, 64x64 per-wave tile (4x4 fragments) -> 16 WMMAs
// per 16 ds_load_b128 per K-step, 16 independent accumulator chains.
// ---------------------------------------------------------------------------

typedef __attribute__((ext_vector_type(16))) __bf16 v16bf;
typedef __attribute__((ext_vector_type(8)))  __bf16 v8bf;
typedef __attribute__((ext_vector_type(8)))  float  v8f;

#define EMBED   1024
#define NHEADS  16
#define HDIM    64
#define BATCH   4
#define SEQ     2048
#define ROWS    (BATCH*SEQ)   // 8192
#define FFND    (2*EMBED)     // 2048

// ---------------------------------------------------------------------------
// WMMA fragment loaders (wave32). ISA 7.12.2 layouts:
//  A (16x32 bf16): lane l -> row M=l&15; K runs [k0+half*8, +8) and
//                  [k0+16+half*8, +8)  where half = l>>4.
//  B (32x16 bf16): lane l -> col N=l&15; K runs [k0+half*8, +8) and
//                  [k0+16+half*8, +8)  (same order as A).
//  C (16x16 f32):  lane l -> col N=l&15; VGPR j -> row M = j + 8*half.
// ---------------------------------------------------------------------------
__device__ __forceinline__ v16bf fragA(const __bf16* base, int stride, int lane) {
  const int l15 = lane & 15, half = lane >> 4;
  const __bf16* p = base + l15 * stride + half * 8;
  v8bf lo = *(const v8bf*)(p);
  v8bf hi = *(const v8bf*)(p + 16);
  v16bf r;
#pragma unroll
  for (int i = 0; i < 8; ++i) { r[i] = lo[i]; r[i + 8] = hi[i]; }
  return r;
}

__device__ __forceinline__ v16bf fragB(const __bf16* base, int stride, int lane) {
  const int l15 = lane & 15, half = lane >> 4;
  const __bf16* p = base + l15 * stride + half * 8;
  v8bf lo = *(const v8bf*)(p);
  v8bf hi = *(const v8bf*)(p + 16);
  v16bf r;
#pragma unroll
  for (int i = 0; i < 8; ++i) { r[i] = lo[i]; r[i + 8] = hi[i]; }
  return r;
}

// ---------------------------------------------------------------------------
// fp32 -> bf16 elementwise convert
// ---------------------------------------------------------------------------
__global__ void cvt_bf16_kernel(const float* __restrict__ s, __bf16* __restrict__ d, int n) {
  int id = blockIdx.x * 256 + threadIdx.x;
  if (id < n) d[id] = (__bf16)s[id];
}

// ---------------------------------------------------------------------------
// fp32 (K x N, row major) -> bf16 transposed (N x K). One-shot weight prep;
// bandwidth-trivial relative to the GEMMs.
// ---------------------------------------------------------------------------
__global__ void transpose_bf16_kernel(const float* __restrict__ W, __bf16* __restrict__ Wt,
                                      int K, int N) {
  int id = blockIdx.x * 256 + threadIdx.x;
  if (id >= K * N) return;
  int k = id / N;
  int n = id - k * N;
  Wt[(size_t)n * K + k] = (__bf16)W[id];
}

// ---------------------------------------------------------------------------
// Tiled bf16 GEMM: C[M][N] = A[M][K] * Bt[N][K]^T (+bias +accum +resid)
// Block tile 128(M) x 256(N), BK=32, 8 waves as 2(row) x 4(col); each wave
// owns a 64x64 tile = 4x4 C fragments. 16 WMMA : 16 ds_load_b128 per K-step.
// M multiple of 128, N multiple of 256, K multiple of 32 (all call sites).
// ---------------------------------------------------------------------------
__global__ __launch_bounds__(256) void gemm_bf16_kernel(
    const __bf16* __restrict__ A, int lda,
    const __bf16* __restrict__ Bt, int ldb,
    const float* __restrict__ bias,
    float* Cf, __bf16* Cb,
    const float* __restrict__ resid,
    int accum, int N, int K)
{
  __shared__ alignas(16) __bf16 sA[128 * 40];   // 40-elem stride: 80B, 16B-aligned
  __shared__ alignas(16) __bf16 sB[256 * 40];

  const int tid = threadIdx.x, lane = tid & 31, wave = tid >> 5;
  const int l15 = lane & 15, half = lane >> 4;
  const int rowBase = blockIdx.y * 128, colBase = blockIdx.x * 256;
  const int wm = wave & 1, wn = wave >> 1;   // 2 row-waves x 4 col-waves

  v8f acc[4][4];
#pragma unroll
  for (int a = 0; a < 4; ++a)
#pragma unroll
    for (int b = 0; b < 4; ++b)
#pragma unroll
      for (int i = 0; i < 8; ++i) acc[a][b][i] = 0.f;

  const int ldRow = tid >> 1;          // 0..127
  const int ldCol = (tid & 1) * 16;    // 0 or 16

  for (int kb = 0; kb < K; kb += 32) {
    __syncthreads();
    {
      const __bf16* ga = A + (size_t)(rowBase + ldRow) * lda + kb + ldCol;
      *(v8bf*)(sA + ldRow * 40 + ldCol)     = *(const v8bf*)(ga);
      *(v8bf*)(sA + ldRow * 40 + ldCol + 8) = *(const v8bf*)(ga + 8);
#pragma unroll
      for (int r = 0; r < 2; ++r) {
        const int br = ldRow + r * 128;
        const __bf16* gb = Bt + (size_t)(colBase + br) * ldb + kb + ldCol;
        *(v8bf*)(sB + br * 40 + ldCol)     = *(const v8bf*)(gb);
        *(v8bf*)(sB + br * 40 + ldCol + 8) = *(const v8bf*)(gb + 8);
      }
    }
    __syncthreads();

    v16bf af[4];
#pragma unroll
    for (int fr = 0; fr < 4; ++fr)
      af[fr] = fragA(sA + (wm * 64 + fr * 16) * 40, 40, lane);
#pragma unroll
    for (int fc = 0; fc < 4; ++fc) {
      v16bf bf = fragB(sB + (wn * 64 + fc * 16) * 40, 40, lane);
#pragma unroll
      for (int fr = 0; fr < 4; ++fr)
        acc[fr][fc] = __builtin_amdgcn_wmma_f32_16x16x32_bf16(
            false, af[fr], false, bf, (short)0, acc[fr][fc], false, false);
    }
  }

#pragma unroll
  for (int fr = 0; fr < 4; ++fr)
#pragma unroll
    for (int fc = 0; fc < 4; ++fc) {
      const int col = colBase + wn * 64 + fc * 16 + l15;
      const float bv = bias ? bias[col] : 0.f;
#pragma unroll
      for (int j = 0; j < 8; ++j) {
        const int row = rowBase + wm * 64 + fr * 16 + j + half * 8;
        const size_t idx = (size_t)row * N + col;
        float v = acc[fr][fc][j] + bv;
        if (accum) v += Cf[idx];
        if (resid) v += resid[idx];
        if (Cf) Cf[idx] = v;
        if (Cb) Cb[idx] = (__bf16)v;
      }
    }
}

// ---------------------------------------------------------------------------
// RoPE + QKV split. qkv (ROWS, 3*EMBED) f32, element (h*64+dd)*3 + {q,k,v}.
// Writes q,k as (b,h,n,hd) bf16 (rotary applied), v transposed (b,h,hd,n) bf16
// so the attention PV matmul's B-operand fragments stream contiguous keys.
// One thread per (b,n,h,pair).
// ---------------------------------------------------------------------------
__global__ void rope_split_kernel(const float* __restrict__ qkv, const float* __restrict__ enc,
                                  __bf16* __restrict__ qb, __bf16* __restrict__ kb,
                                  __bf16* __restrict__ vtb) {
  const int idx = blockIdx.x * 256 + threadIdx.x;  // 4*2048*16*32 threads
  const int pr = idx & 31;
  const int h  = (idx >> 5) & 15;
  const int n  = (idx >> 9) & 2047;
  const int b  = idx >> 20;
  if (b >= BATCH) return;
  const int d0 = pr * 2;
  const float* base = qkv + (size_t)(b * SEQ + n) * 3 * EMBED + (h * HDIM + d0) * 3;
  const float q0 = base[0], k0 = base[1], v0 = base[2];
  const float q1 = base[3], k1 = base[4], v1 = base[5];
  const float f00 = enc[(size_t)n * HDIM + d0];
  const float f01 = enc[(size_t)n * HDIM + d0 + 1];
  const float f10 = enc[(size_t)SEQ * HDIM + (size_t)n * HDIM + d0];
  const float f11 = enc[(size_t)SEQ * HDIM + (size_t)n * HDIM + d0 + 1];
  // out[2i] = t[2i]*f0[2i] - t[2i+1]*f1[2i]; out[2i+1] = t[2i+1]*f0[2i+1] + t[2i]*f1[2i+1]
  const float qo0 = q0 * f00 - q1 * f10;
  const float qo1 = q1 * f01 + q0 * f11;
  const float ko0 = k0 * f00 - k1 * f10;
  const float ko1 = k1 * f01 + k0 * f11;
  const size_t o = ((size_t)(b * NHEADS + h) * SEQ + n) * HDIM + d0;
  qb[o] = (__bf16)qo0; qb[o + 1] = (__bf16)qo1;
  kb[o] = (__bf16)ko0; kb[o + 1] = (__bf16)ko1;
  const size_t ov = ((size_t)(b * NHEADS + h) * HDIM + d0) * SEQ + n;
  vtb[ov] = (__bf16)v0; vtb[ov + SEQ] = (__bf16)v1;
}

// ---------------------------------------------------------------------------
// Flash attention. grid = (B*H, SEQ/128). Block: 128 q rows, 8 waves; wave w
// owns q rows [w*16, w*16+16). Online softmax; exp/max in fp32 with cross-lane
// shfl_xor reductions over the 16-lane half groups (matches C-layout rows).
// P re-laid-out through per-wave LDS (aliases K tile; barrier-protected).
// ---------------------------------------------------------------------------
__global__ __launch_bounds__(256) void attn_kernel(
    const __bf16* __restrict__ qb, const __bf16* __restrict__ kb,
    const __bf16* __restrict__ vtb, __bf16* __restrict__ msg)
{
  __shared__ alignas(16) __bf16 sKP[8 * 16 * 136];  // union: K tile [128][72] | P [wave][16][136]
  __shared__ alignas(16) __bf16 sV[64 * 136];       // V^T tile [hd][key]

  const int bh = blockIdx.x, qt = blockIdx.y;
  const int b = bh / NHEADS, h = bh % NHEADS;
  const __bf16* Q  = qb  + ((size_t)bh * SEQ + qt * 128) * HDIM;
  const __bf16* Kp = kb  + (size_t)bh * SEQ * HDIM;
  const __bf16* Vt = vtb + (size_t)bh * HDIM * SEQ;

  const int tid = threadIdx.x, lane = tid & 31, wave = tid >> 5;
  const int l15 = lane & 15, half = lane >> 4;

  // Q fragments: per-wave, K=64 -> 2 A-fragments, loaded once from global.
  v16bf qf[2];
#pragma unroll
  for (int kk = 0; kk < 2; ++kk)
    qf[kk] = fragA(Q + (size_t)wave * 16 * HDIM + kk * 32, HDIM, lane);

  float m[8], l[8];
  v8f ctx[4];
#pragma unroll
  for (int j = 0; j < 8; ++j) { m[j] = -1e30f; l[j] = 0.f; }
#pragma unroll
  for (int f = 0; f < 4; ++f)
#pragma unroll
    for (int j = 0; j < 8; ++j) ctx[f][j] = 0.f;

  const float scale = 0.125f;  // 1/sqrt(64)

  for (int kt = 0; kt < SEQ / 128; ++kt) {
    __syncthreads();  // previous iteration's P/V reads complete
    // cooperative load: K tile (128 keys x 64, stride 72) and V^T tile (64 x 128, stride 136)
    {
      const __bf16* Kt = Kp + (size_t)kt * 128 * HDIM;
#pragma unroll
      for (int c = 0; c < 4; ++c) {
        int ch = tid * 4 + c;
        int row = ch >> 3, co = (ch & 7) * 8;
        *(v8bf*)(sKP + row * 72 + co) = *(const v8bf*)(Kt + (size_t)row * HDIM + co);
      }
      const __bf16* Vg = Vt + kt * 128;
#pragma unroll
      for (int c = 0; c < 4; ++c) {
        int ch = tid * 4 + c;
        int row = ch >> 4, co = (ch & 15) * 8;
        *(v8bf*)(sV + row * 136 + co) = *(const v8bf*)(Vg + (size_t)row * SEQ + co);
      }
    }
    __syncthreads();

    // S = Q K^T : 8 column fragments of 16x16, K=64 -> 2 WMMA each
    v8f S[8];
#pragma unroll
    for (int f = 0; f < 8; ++f) {
      v8f a;
#pragma unroll
      for (int j = 0; j < 8; ++j) a[j] = 0.f;
#pragma unroll
      for (int kk = 0; kk < 2; ++kk) {
        v16bf bf = fragB(sKP + (f * 16) * 72 + kk * 32, 72, lane);
        a = __builtin_amdgcn_wmma_f32_16x16x32_bf16(false, qf[kk], false, bf,
                                                    (short)0, a, false, false);
      }
      S[f] = a;
    }

    // online softmax (lane holds 8 rows j -> row = j + 8*half, col = l15 + 16*f)
    float alpha[8];
#pragma unroll
    for (int j = 0; j < 8; ++j) {
      float mx = -1e30f;
#pragma unroll
      for (int f = 0; f < 8; ++f) { float v = S[f][j] * scale; S[f][j] = v; mx = fmaxf(mx, v); }
      for (int msk = 1; msk < 16; msk <<= 1) mx = fmaxf(mx, __shfl_xor(mx, msk, 32));
      const float nm = fmaxf(m[j], mx);
      alpha[j] = __expf(m[j] - nm);
      m[j] = nm;
      float rs = 0.f;
#pragma unroll
      for (int f = 0; f < 8; ++f) { float p = __expf(S[f][j] - nm); S[f][j] = p; rs += p; }
      for (int msk = 1; msk < 16; msk <<= 1) rs += __shfl_xor(rs, msk, 32);
      l[j] = l[j] * alpha[j] + rs;
    }
#pragma unroll
    for (int f = 0; f < 4; ++f)
#pragma unroll
      for (int j = 0; j < 8; ++j) ctx[f][j] *= alpha[j];

    __syncthreads();  // all waves finished reading K tile before P overwrites it
    __bf16* P = sKP + wave * 16 * 136;
#pragma unroll
    for (int f = 0; f < 8; ++f)
#pragma unroll
      for (int j = 0; j < 8; ++j)
        P[(j + half * 8) * 136 + f * 16 + l15] = (__bf16)S[f][j];
    __syncthreads();

    // ctx += P (16x128) @ V (128x64): 4 output frags x 4 K-steps
#pragma unroll
    for (int f = 0; f < 4; ++f) {
      v8f a = ctx[f];
#pragma unroll
      for (int kk = 0; kk < 4; ++kk) {
        v16bf af = fragA(P + kk * 32, 136, lane);
        v16bf bf = fragB(sV + (f * 16) * 136 + kk * 32, 136, lane);
        a = __builtin_amdgcn_wmma_f32_16x16x32_bf16(false, af, false, bf,
                                                    (short)0, a, false, false);
      }
      ctx[f] = a;
    }
  }

  // finalize: context / l -> message rows (b, n, EMBED) bf16
#pragma unroll
  for (int f = 0; f < 4; ++f)
#pragma unroll
    for (int j = 0; j < 8; ++j) {
      const int n = qt * 128 + wave * 16 + j + half * 8;
      const float v = ctx[f][j] / l[j];
      msg[((size_t)b * SEQ + n) * EMBED + h * HDIM + f * 16 + l15] = (__bf16)v;
    }
}

// ---------------------------------------------------------------------------
// LayerNorm (2048 wide) + exact GELU, fp32 in -> bf16 out. One block per row.
// ---------------------------------------------------------------------------
__global__ __launch_bounds__(256) void ln_gelu_kernel(
    const float* __restrict__ H, const float* __restrict__ sc,
    const float* __restrict__ bi, __bf16* __restrict__ Hb)
{
  __shared__ float red[256], red2[256];
  const int row = blockIdx.x, tid = threadIdx.x;
  const float* hr = H + (size_t)row * FFND;
  float v[8];
  float s = 0.f, s2 = 0.f;
#pragma unroll
  for (int i = 0; i < 8; ++i) {
    v[i] = hr[tid + i * 256];
    s += v[i]; s2 += v[i] * v[i];
  }
  red[tid] = s; red2[tid] = s2;
  __syncthreads();
  for (int st = 128; st > 0; st >>= 1) {
    if (tid < st) { red[tid] += red[tid + st]; red2[tid] += red2[tid + st]; }
    __syncthreads();
  }
  const float mu = red[0] * (1.f / FFND);
  const float var = red2[0] * (1.f / FFND) - mu * mu;
  const float rs = rsqrtf(var + 1e-5f);
#pragma unroll
  for (int i = 0; i < 8; ++i) {
    const int c = tid + i * 256;
    const float y = (v[i] - mu) * rs * sc[c] + bi[c];
    const float g = 0.5f * y * (1.f + erff(y * 0.70710678118f));
    Hb[(size_t)row * FFND + c] = (__bf16)g;
  }
}

// ---------------------------------------------------------------------------
// Host orchestration
// ---------------------------------------------------------------------------
extern "C" void kernel_launch(void* const* d_in, const int* in_sizes, int n_in,
                              void* d_out, int out_size, void* d_ws, size_t ws_size,
                              hipStream_t stream) {
  const float* x       = (const float*)d_in[0];
  const float* enc     = (const float*)d_in[1];
  const float* Wqkv_w  = (const float*)d_in[2];
  const float* Wqkv_b  = (const float*)d_in[3];
  const float* out_w   = (const float*)d_in[4];
  const float* out_b   = (const float*)d_in[5];
  const float* ffn0_w  = (const float*)d_in[6];
  const float* ffn0_b  = (const float*)d_in[7];
  const float* ln_s    = (const float*)d_in[8];
  const float* ln_b    = (const float*)d_in[9];
  const float* ffn3_w  = (const float*)d_in[10];
  const float* ffn3_b  = (const float*)d_in[11];
  float* out = (float*)d_out;

  // workspace carve (~323 MB), 256B aligned slices
  size_t off = 0;
  auto carve = [&](size_t bytes) -> void* {
    void* p = (char*)d_ws + off;
    off += (bytes + 255) & ~(size_t)255;
    return p;
  };
  __bf16* xbf    = (__bf16*)carve((size_t)ROWS * EMBED * 2);
  __bf16* WqkvT  = (__bf16*)carve((size_t)3 * EMBED * EMBED * 2);
  __bf16* outT   = (__bf16*)carve((size_t)EMBED * EMBED * 2);
  __bf16* ffn0T  = (__bf16*)carve((size_t)FFND * FFND * 2);
  __bf16* ffn3T  = (__bf16*)carve((size_t)EMBED * FFND * 2);
  float*  qkv    = (float*) carve((size_t)ROWS * 3 * EMBED * 4);
  __bf16* q_bf   = (__bf16*)carve((size_t)ROWS * EMBED * 2);
  __bf16* k_bf   = (__bf16*)carve((size_t)ROWS * EMBED * 2);
  __bf16* vt_bf  = (__bf16*)carve((size_t)ROWS * EMBED * 2);
  __bf16* msg_bf = (__bf16*)carve((size_t)ROWS * EMBED * 2);
  __bf16* msg2bf = (__bf16*)carve((size_t)ROWS * EMBED * 2);
  float*  h_f32  = (float*) carve((size_t)ROWS * FFND * 4);
  __bf16* h_bf   = (__bf16*)carve((size_t)ROWS * FFND * 2);

  // 1) precision prep: x -> bf16; weights -> transposed bf16 [N][K]
  {
    int n = ROWS * EMBED;
    cvt_bf16_kernel<<<(n + 255) / 256, 256, 0, stream>>>(x, xbf, n);
  }
  transpose_bf16_kernel<<<(EMBED * 3 * EMBED + 255) / 256, 256, 0, stream>>>(Wqkv_w, WqkvT, EMBED, 3 * EMBED);
  transpose_bf16_kernel<<<(EMBED * EMBED + 255) / 256, 256, 0, stream>>>(out_w, outT, EMBED, EMBED);
  transpose_bf16_kernel<<<(FFND * FFND + 255) / 256, 256, 0, stream>>>(ffn0_w, ffn0T, FFND, FFND);
  transpose_bf16_kernel<<<(FFND * EMBED + 255) / 256, 256, 0, stream>>>(ffn3_w, ffn3T, FFND, EMBED);

  // 2) QKV projection: qkv = x @ Wqkv + b   (8192x3072, K=1024)
  gemm_bf16_kernel<<<dim3(3 * EMBED / 256, ROWS / 128), 256, 0, stream>>>(
      xbf, EMBED, WqkvT, EMBED, Wqkv_b, qkv, nullptr, nullptr, 0, 3 * EMBED, EMBED);

  // 3) RoPE + split into q/k (b,h,n,hd) and v^T (b,h,hd,n)
  {
    int n = BATCH * SEQ * NHEADS * (HDIM / 2);
    rope_split_kernel<<<(n + 255) / 256, 256, 0, stream>>>(qkv, enc, q_bf, k_bf, vt_bf);
  }

  // 4) attention -> message (b,n,EMBED) bf16
  attn_kernel<<<dim3(BATCH * NHEADS, SEQ / 128), 256, 0, stream>>>(q_bf, k_bf, vt_bf, msg_bf);

  // 5) out-proj: message2 = message @ out_w + out_b  (bf16 out)
  gemm_bf16_kernel<<<dim3(EMBED / 256, ROWS / 128), 256, 0, stream>>>(
      msg_bf, EMBED, outT, EMBED, out_b, nullptr, msg2bf, nullptr, 0, EMBED, EMBED);

  // 6) ffn0 over concat([x, message2]): split into two GEMMs into h_f32
  gemm_bf16_kernel<<<dim3(FFND / 256, ROWS / 128), 256, 0, stream>>>(
      xbf, EMBED, ffn0T, FFND, ffn0_b, h_f32, nullptr, nullptr, 0, FFND, EMBED);
  gemm_bf16_kernel<<<dim3(FFND / 256, ROWS / 128), 256, 0, stream>>>(
      msg2bf, EMBED, ffn0T + EMBED /* k-offset 1024 */, FFND, nullptr, h_f32, nullptr, nullptr,
      1, FFND, EMBED);

  // 7) LayerNorm + exact GELU -> h_bf
  ln_gelu_kernel<<<ROWS, 256, 0, stream>>>(h_f32, ln_s, ln_b, h_bf);

  // 8) ffn3 + bias + residual x -> d_out (fp32)
  gemm_bf16_kernel<<<dim3(EMBED / 256, ROWS / 128), 256, 0, stream>>>(
      h_bf, FFND, ffn3T, FFND, ffn3_b, out, nullptr, x, 0, EMBED, FFND);

  (void)in_sizes; (void)n_in; (void)out_size; (void)ws_size;
}
